// CovClassifier_59115929862234
// MI455X (gfx1250) — matrix operand order repaired and checked by reading, not compiled, not run on hardware
//
#include <hip/hip_runtime.h>

typedef __bf16 bf16_t;
typedef __attribute__((ext_vector_type(16))) __bf16 v16bf;
typedef __attribute__((ext_vector_type(8)))  float  v8f;
typedef int v4i __attribute__((__vector_size__(16)));   // matches async builtin param type

#define CDIM 256
#define MDIM 196
#define BDIM 256
#define DTRIU 32896

// gfx1250 async global->LDS path, guarded so the source compiles either way.
#if defined(__gfx1250__) && \
    __has_builtin(__builtin_amdgcn_global_load_async_to_lds_b128) && \
    __has_builtin(__builtin_amdgcn_s_wait_asynccnt)
#define USE_ASYNC 1
#else
#define USE_ASYNC 0
#endif

// ---------------------------------------------------------------------------
// stats: per (b,c) row mean and centered sum-of-squares / M (trace pieces)
// ---------------------------------------------------------------------------
__global__ __launch_bounds__(256)
void stats_kernel(const float* __restrict__ x, float* __restrict__ mean,
                  float* __restrict__ rsum)
{
    int row  = blockIdx.x * 8 + (threadIdx.x >> 5);   // [0, B*C)
    int lane = threadIdx.x & 31;
    const float* xr = x + (size_t)row * MDIM;
    float s1 = 0.f, s2 = 0.f;
    for (int m = lane; m < MDIM; m += 32) { float v = xr[m]; s1 += v; s2 += v * v; }
    #pragma unroll
    for (int off = 16; off > 0; off >>= 1) {
        s1 += __shfl_xor(s1, off, 32);
        s2 += __shfl_xor(s2, off, 32);
    }
    if (lane == 0) {
        float mu = s1 * (1.f / MDIM);
        mean[row] = mu;
        rsum[row] = (s2 - (float)MDIM * mu * mu) * (1.f / MDIM);
    }
}

// ---------------------------------------------------------------------------
// trace reduction: normA[b] = sum_c rsum[b,c]  (deterministic LDS tree)
// ---------------------------------------------------------------------------
__global__ __launch_bounds__(256)
void norm_kernel(const float* __restrict__ rsum, float* __restrict__ normA)
{
    __shared__ float sm[256];
    int b = blockIdx.x, tid = threadIdx.x;
    sm[tid] = rsum[b * CDIM + tid];
    __syncthreads();
    #pragma unroll
    for (int s = 128; s > 0; s >>= 1) {
        if (tid < s) sm[tid] += sm[tid + s];
        __syncthreads();
    }
    if (tid == 0) normA[b] = sm[0];
}

// ---------------------------------------------------------------------------
// covariance via WMMA: Ahat = (xc xc^T) / (M * trace), ZY = 0.5*(3I - Ahat)
// 64x64 tile per block, 8 waves x (16x32) per wave, K padded 196 -> 224
// ---------------------------------------------------------------------------
__global__ __launch_bounds__(256)
void cov_kernel(const float* __restrict__ x, const float* __restrict__ mean,
                const float* __restrict__ normA,
                bf16_t* __restrict__ Ahat, bf16_t* __restrict__ ZY)
{
    const int b  = blockIdx.z;
    const int m0 = blockIdx.y * 64;
    const int n0 = blockIdx.x * 64;

    __shared__ __align__(16) bf16_t As[64][32];
    __shared__ __align__(16) bf16_t Bs[64][32];

    const int tid  = threadIdx.x;
    const int wave = tid >> 5, lane = tid & 31;
    const int g = lane >> 4, hl = lane & 15;
    const int wm = (wave >> 1) * 16, wn = (wave & 1) * 32;

    const int e  = tid * 8;
    const int sr = e >> 5;    // 0..63
    const int sc = e & 31;    // 0,8,16,24

    v8f acc0 = {}; v8f acc1 = {};

    for (int k0 = 0; k0 < 224; k0 += 32) {
        __syncthreads();
        {
            int row  = m0 + sr;
            float mu = mean[b * CDIM + row];
            const float* xp = x + ((size_t)b * CDIM + row) * MDIM;
            #pragma unroll
            for (int i = 0; i < 8; ++i) {
                int k = k0 + sc + i;
                As[sr][sc + i] = (k < MDIM) ? (bf16_t)(xp[k] - mu) : (bf16_t)0.f;
            }
            row = n0 + sr;
            mu  = mean[b * CDIM + row];
            xp  = x + ((size_t)b * CDIM + row) * MDIM;
            #pragma unroll
            for (int i = 0; i < 8; ++i) {
                int k = k0 + sc + i;
                Bs[sr][sc + i] = (k < MDIM) ? (bf16_t)(xp[k] - mu) : (bf16_t)0.f;
            }
        }
        __syncthreads();

        v16bf af, bf0, bf1;
        #pragma unroll
        for (int i = 0; i < 8; ++i) {
            af[i]     = As[wm + hl][8 * g + i];
            af[8 + i] = As[wm + hl][16 + 8 * g + i];
        }
        #pragma unroll
        for (int i = 0; i < 8; ++i) {
            bf0[i]     = Bs[wn + hl][8 * g + i];
            bf0[8 + i] = Bs[wn + hl][16 + 8 * g + i];
            bf1[i]     = Bs[wn + 16 + hl][8 * g + i];
            bf1[8 + i] = Bs[wn + 16 + hl][16 + 8 * g + i];
        }
        acc0 = __builtin_amdgcn_wmma_f32_16x16x32_bf16(false, af, false, bf0, (short)0, acc0, false, false);
        acc1 = __builtin_amdgcn_wmma_f32_16x16x32_bf16(false, af, false, bf1, (short)0, acc1, false, false);
    }

    const float inv = 1.f / ((float)MDIM * normA[b]);
    bf16_t* Ab = Ahat + (size_t)b * CDIM * CDIM;
    bf16_t* Zb = ZY   + (size_t)b * CDIM * CDIM;
    #pragma unroll
    for (int r = 0; r < 8; ++r) {
        int row  = m0 + wm + 8 * g + r;
        int col0 = n0 + wn + hl;
        float v0 = acc0[r] * inv;
        float v1 = acc1[r] * inv;
        Ab[row * CDIM + col0]      = (bf16_t)v0;
        Ab[row * CDIM + col0 + 16] = (bf16_t)v1;
        Zb[row * CDIM + col0]      = (bf16_t)(0.5f * ((row == col0      ? 3.f : 0.f) - v0));
        Zb[row * CDIM + col0 + 16] = (bf16_t)(0.5f * ((row == col0 + 16 ? 3.f : 0.f) - v1));
    }
}

// ---------------------------------------------------------------------------
// batched 256x256x256 GEMM computing R = P @ Q^T (all NS operands are
// polynomials in Ahat => symmetric, so P@Q^T == P@Q up to bf16 rounding).
// Row-major staging for BOTH panels (no transpose scatter).
// Block tile 128x64, 8 waves, each wave 32x32 (4 WMMA accumulators).
// MODE 0: R = P@Qt      MODE 1: R = 0.5*(3I - P@Qt)   MODE 2: R = (P@Qt)*sqrt(normA)
// ---------------------------------------------------------------------------
template<int MODE>
__global__ __launch_bounds__(256)
void gemm_ns(const bf16_t* __restrict__ P, const bf16_t* __restrict__ Q,
             bf16_t* __restrict__ R, const float* __restrict__ normA)
{
    const int b  = blockIdx.z;
    const int m0 = blockIdx.y * 128;
    const int n0 = blockIdx.x * 64;
    const bf16_t* Pb = P + (size_t)b * CDIM * CDIM;
    const bf16_t* Qb = Q + (size_t)b * CDIM * CDIM;
    bf16_t*       Rb = R + (size_t)b * CDIM * CDIM;

    __shared__ __align__(16) bf16_t As[2][128][32];
    __shared__ __align__(16) bf16_t Bs[2][64][32];

    const int tid  = threadIdx.x;
    const int wave = tid >> 5, lane = tid & 31;
    const int g = lane >> 4, hl = lane & 15;
    const int wm = (wave >> 1) * 32, wn = (wave & 1) * 32;

    const int sr = tid >> 2;          // 0..63
    const int sc = (tid & 3) * 8;     // 0,8,16,24

    const bf16_t* pa0 = Pb + (m0 + sr) * CDIM + sc;        // A rows 0..63
    const bf16_t* pa1 = Pb + (m0 + 64 + sr) * CDIM + sc;   // A rows 64..127
    const bf16_t* pq  = Qb + (n0 + sr) * CDIM + sc;        // Q row n (acts as B^T)

    v8f acc[2][2] = {};

    auto compute = [&](int cur) {
        v16bf a0, a1, q0, q1;
        #pragma unroll
        for (int i = 0; i < 8; ++i) {
            a0[i]     = As[cur][wm + hl][8 * g + i];
            a0[8 + i] = As[cur][wm + hl][16 + 8 * g + i];
            a1[i]     = As[cur][wm + 16 + hl][8 * g + i];
            a1[8 + i] = As[cur][wm + 16 + hl][16 + 8 * g + i];
        }
        #pragma unroll
        for (int i = 0; i < 8; ++i) {
            q0[i]     = Bs[cur][wn + hl][8 * g + i];
            q0[8 + i] = Bs[cur][wn + hl][16 + 8 * g + i];
            q1[i]     = Bs[cur][wn + 16 + hl][8 * g + i];
            q1[8 + i] = Bs[cur][wn + 16 + hl][16 + 8 * g + i];
        }
        acc[0][0] = __builtin_amdgcn_wmma_f32_16x16x32_bf16(false, a0, false, q0, (short)0, acc[0][0], false, false);
        acc[0][1] = __builtin_amdgcn_wmma_f32_16x16x32_bf16(false, a0, false, q1, (short)0, acc[0][1], false, false);
        acc[1][0] = __builtin_amdgcn_wmma_f32_16x16x32_bf16(false, a1, false, q0, (short)0, acc[1][0], false, false);
        acc[1][1] = __builtin_amdgcn_wmma_f32_16x16x32_bf16(false, a1, false, q1, (short)0, acc[1][1], false, false);
    };

#if USE_ASYNC
    // double-buffered async global->LDS pipeline (ASYNCcnt-tracked)
    auto issue = [&](int bufi, int k0) {
        __builtin_amdgcn_global_load_async_to_lds_b128(
            (v4i*)(pa0 + k0), (v4i*)&As[bufi][sr][sc], 0, 0);
        __builtin_amdgcn_global_load_async_to_lds_b128(
            (v4i*)(pa1 + k0), (v4i*)&As[bufi][64 + sr][sc], 0, 0);
        __builtin_amdgcn_global_load_async_to_lds_b128(
            (v4i*)(pq + k0), (v4i*)&Bs[bufi][sr][sc], 0, 0);
    };
    issue(0, 0);
    for (int i = 0; i < 8; ++i) {
        const int cur = i & 1;
        if (i < 7) {
            issue(cur ^ 1, (i + 1) * 32);          // prefetch next panel
            __builtin_amdgcn_s_wait_asynccnt(3);   // current panel landed
        } else {
            __builtin_amdgcn_s_wait_asynccnt(0);
        }
        __syncthreads();
        compute(cur);
        __syncthreads();                           // reads done before buffer reuse
    }
#else
    // register-staged software pipeline: next panel's global loads are issued
    // before computing the current panel, so vmem overlaps WMMA.
    uint4 ra0 = *(const uint4*)pa0;
    uint4 ra1 = *(const uint4*)pa1;
    uint4 rq  = *(const uint4*)pq;
    for (int i = 0; i < 8; ++i) {
        const int cur = i & 1;
        *(uint4*)&As[cur][sr][sc]      = ra0;
        *(uint4*)&As[cur][64 + sr][sc] = ra1;
        *(uint4*)&Bs[cur][sr][sc]      = rq;
        __syncthreads();
        if (i < 7) {
            int k = (i + 1) * 32;
            ra0 = *(const uint4*)(pa0 + k);
            ra1 = *(const uint4*)(pa1 + k);
            rq  = *(const uint4*)(pq + k);
        }
        compute(cur);
        __syncthreads();
    }
#endif

    float s = 1.f;
    if (MODE == 2) s = sqrtf(normA[b]);
    #pragma unroll
    for (int ti = 0; ti < 2; ++ti) {
        #pragma unroll
        for (int tj = 0; tj < 2; ++tj) {
            #pragma unroll
            for (int r = 0; r < 8; ++r) {
                int row = m0 + wm + 16 * ti + 8 * g + r;
                int col = n0 + wn + 16 * tj + hl;
                float v = acc[ti][tj][r];
                if (MODE == 1)      v = 0.5f * ((row == col ? 3.f : 0.f) - v);
                else if (MODE == 2) v *= s;
                Rb[row * CDIM + col] = (bf16_t)v;
            }
        }
    }
}

// ---------------------------------------------------------------------------
// triu extract + 2-way FC, one block per batch, deterministic LDS tree reduce
// ---------------------------------------------------------------------------
__global__ __launch_bounds__(256)
void triufc_kernel(const bf16_t* __restrict__ Y, const float* __restrict__ fcw,
                   const float* __restrict__ fcb, float* __restrict__ out)
{
    const int b = blockIdx.x, tid = threadIdx.x;
    const bf16_t* Yb = Y + (size_t)b * CDIM * CDIM;
    const float* w0 = fcw;
    const float* w1 = fcw + DTRIU;
    float a0 = 0.f, a1 = 0.f;
    for (int e = tid; e < CDIM * CDIM; e += 256) {
        int i = e >> 8, j = e & 255;
        if (j < i) continue;
        float y = (float)Yb[e];
        int t = i * CDIM - (i * (i - 1)) / 2 + (j - i);
        a0 += y * w0[t];
        a1 += y * w1[t];
    }
    __shared__ float s0[256], s1[256];
    s0[tid] = a0; s1[tid] = a1;
    __syncthreads();
    #pragma unroll
    for (int s = 128; s > 0; s >>= 1) {
        if (tid < s) { s0[tid] += s0[tid + s]; s1[tid] += s1[tid + s]; }
        __syncthreads();
    }
    if (tid == 0) {
        out[b * 2 + 0] = s0[0] + fcb[0];
        out[b * 2 + 1] = s1[0] + fcb[1];
    }
}

// ---------------------------------------------------------------------------
extern "C" void kernel_launch(void* const* d_in, const int* in_sizes, int n_in,
                              void* d_out, int out_size, void* d_ws, size_t ws_size,
                              hipStream_t stream)
{
    const float* x   = (const float*)d_in[0];
    const float* fcw = (const float*)d_in[1];
    const float* fcb = (const float*)d_in[2];
    float* out = (float*)d_out;

    const size_t NMAT = (size_t)BDIM * CDIM * CDIM;   // elements per matrix buf
    char* w = (char*)d_ws;
    bf16_t* buf0 = (bf16_t*)(w + 0 * NMAT * 2);
    bf16_t* buf1 = (bf16_t*)(w + 1 * NMAT * 2);
    bf16_t* buf2 = (bf16_t*)(w + 2 * NMAT * 2);
    bf16_t* buf3 = (bf16_t*)(w + 3 * NMAT * 2);
    float*  mean  = (float*)(w + 4 * NMAT * 2);
    float*  rsum  = mean + (size_t)BDIM * CDIM;
    float*  normA = rsum + (size_t)BDIM * CDIM;

    stats_kernel<<<dim3(BDIM * CDIM / 8), dim3(256), 0, stream>>>(x, mean, rsum);
    norm_kernel <<<dim3(BDIM),            dim3(256), 0, stream>>>(rsum, normA);

    // buf0 = Ahat, buf1 = ZY = 0.5*(3I - Ahat)
    cov_kernel<<<dim3(4, 4, BDIM), dim3(256), 0, stream>>>(x, mean, normA, buf0, buf1);

    dim3 gg(4, 2, BDIM), bb(256);   // 64-wide N tiles, 128-tall M tiles

    // Y = Ahat @ ZY ; Z = ZY
    gemm_ns<0><<<gg, bb, 0, stream>>>(buf0, buf1, buf2, normA);
    bf16_t* Y = buf2; bf16_t* Z = buf1; bf16_t* f0 = buf0; bf16_t* f1 = buf3;

    for (int it = 0; it < 3; ++it) {
        bf16_t* T = f0;
        gemm_ns<1><<<gg, bb, 0, stream>>>(Z, Y, T, normA);   // T  = 0.5*(3I - Z@Y)
        bf16_t* Yn = f1;
        gemm_ns<0><<<gg, bb, 0, stream>>>(Y, T, Yn, normA);  // Yn = Y@T
        bf16_t* Zn = Y;                                      // old Y slot now free
        gemm_ns<0><<<gg, bb, 0, stream>>>(T, Z, Zn, normA);  // Zn = T@Z
        f0 = Z; f1 = T;
        Y = Yn; Z = Zn;
    }

    bf16_t* T = f0;
    gemm_ns<1><<<gg, bb, 0, stream>>>(Z, Y, T, normA);       // T = 0.5*(3I - Z@Y)
    bf16_t* Yf = f1;
    gemm_ns<2><<<gg, bb, 0, stream>>>(Y, T, Yf, normA);      // Yf = (Y@T)*sqrt(tr)

    triufc_kernel<<<dim3(BDIM), dim3(256), 0, stream>>>(Yf, fcw, fcb, out);
}
